// VectorizedMoELayer_64244120814228
// MI455X (gfx1250) — compile-verified, matrix operand order checked by reading, not run on hardware
//
#include <hip/hip_runtime.h>

// ---------- CDNA5 vector types ----------
typedef __attribute__((ext_vector_type(16))) __bf16        v16bf;
typedef __attribute__((ext_vector_type(8)))  float         v8f;
typedef __attribute__((ext_vector_type(4)))  float         f32x4;
typedef __attribute__((ext_vector_type(4)))  unsigned int  u32x4;
typedef __attribute__((ext_vector_type(4)))  int           i32x4;
typedef __attribute__((ext_vector_type(8)))  int           i32x8;

// ---------- problem dims ----------
#define T_TOK 8192
#define DDIM  1024
#define HDIM  2048
#define NEXP  8
#define TOPKK 2
#define NSLOT (T_TOK * TOPKK)

// ---------- GEMM tiling (wave32, 8 waves/block) ----------
#define BM  128
#define BN  128
#define BK  32
#define BKP 40   // padded LDS row (ushort units) -> 80B rows, 16B-aligned frags

#if defined(__has_builtin)
#if __has_builtin(__builtin_amdgcn_tensor_load_to_lds) && __has_builtin(__builtin_amdgcn_s_wait_tensorcnt)
#define HAVE_TDM 1
#endif
#endif

__device__ __forceinline__ unsigned short f2bf(float f) {
    union { __bf16 b; unsigned short u; } c;
    c.b = (__bf16)f;                       // RNE on CDNA5
    return c.u;
}

// 16-bit WMMA operand fragment from padded LDS tile:
// lanes 0-15: row = sub, k = 0..7 & 16..23 ; lanes 16-31: row = sub, k = 8..15 & 24..31
__device__ __forceinline__ const unsigned short* frag_ptr(const unsigned short* tile, int subrow0) {
    int lane = threadIdx.x & 31;
    int r  = subrow0 + (lane & 15);
    int kb = (lane & 16) ? 8 : 0;
    return tile + r * BKP + kb;
}

__device__ __forceinline__ v16bf load_frag(const unsigned short* p) {
    union { v16bf v; u32x4 q[2]; } u;
    u.q[0] = *(const u32x4*)(p);        // k 0..7   (16B aligned)
    u.q[1] = *(const u32x4*)(p + 16);   // k 16..23
    return u.v;
}

// =====================================================================
// 1. Router: one wave32 per token
// =====================================================================
__global__ __launch_bounds__(256)
void moe_router(const float* __restrict__ x, const float* __restrict__ rw,
                int* __restrict__ counts, int* __restrict__ expert_idx,
                int* __restrict__ rankb, float* __restrict__ coef2,
                float* __restrict__ prob)
{
    __shared__ float lrw[NEXP * DDIM];   // 32 KB of router weights
    int tid = threadIdx.x;
    for (int i = tid; i < NEXP * DDIM; i += 256) lrw[i] = rw[i];
    __syncthreads();

    int wave = tid >> 5, lane = tid & 31;
    int t = blockIdx.x * 8 + wave;

    float acc[NEXP];
#pragma unroll
    for (int e = 0; e < NEXP; ++e) acc[e] = 0.f;

    const float* xr = x + (size_t)t * DDIM;
    for (int d = lane; d < DDIM; d += 32) {
        float xv = xr[d];
#pragma unroll
        for (int e = 0; e < NEXP; ++e) acc[e] += xv * lrw[e * DDIM + d];
    }
#pragma unroll
    for (int e = 0; e < NEXP; ++e) {
#pragma unroll
        for (int off = 16; off > 0; off >>= 1)
            acc[e] += __shfl_xor(acc[e], off, 32);
    }

    if (lane == 0) {
        // full softmax (for aux-loss P)
        float mx = acc[0];
#pragma unroll
        for (int e = 1; e < NEXP; ++e) mx = fmaxf(mx, acc[e]);
        float p[NEXP], s = 0.f;
#pragma unroll
        for (int e = 0; e < NEXP; ++e) { p[e] = __expf(acc[e] - mx); s += p[e]; }
        float inv = 1.f / s;
#pragma unroll
        for (int e = 0; e < NEXP; ++e) prob[(size_t)t * NEXP + e] = p[e] * inv;

        // top-2 + softmax over the pair
        int i0 = 0;
#pragma unroll
        for (int e = 1; e < NEXP; ++e) if (acc[e] > acc[i0]) i0 = e;
        int i1 = (i0 == 0) ? 1 : 0;
#pragma unroll
        for (int e = 0; e < NEXP; ++e) if (e != i0 && acc[e] > acc[i1]) i1 = e;
        float m2 = fmaxf(acc[i0], acc[i1]);
        float w0 = __expf(acc[i0] - m2), w1 = __expf(acc[i1] - m2);
        float wsum = w0 + w1; w0 /= wsum; w1 /= wsum;

        int r0 = atomicAdd(&counts[i0], 1);
        int r1 = atomicAdd(&counts[i1], 1);
        expert_idx[t * 2 + 0] = i0;  expert_idx[t * 2 + 1] = i1;
        rankb[t * 2 + 0] = r0;       rankb[t * 2 + 1] = r1;
        coef2[t * 2 + 0] = w0;       coef2[t * 2 + 1] = w1;
    }
}

// =====================================================================
// 2. Deterministic prob reduction: one block per expert
// =====================================================================
__global__ __launch_bounds__(256)
void moe_reduce_prob(const float* __restrict__ prob, float* __restrict__ sum_prob)
{
    __shared__ float red[256];
    int e = blockIdx.x, tid = threadIdx.x;
    float s = 0.f;
    for (int t = tid; t < T_TOK; t += 256) s += prob[(size_t)t * NEXP + e];
    red[tid] = s;
    __syncthreads();
    for (int w = 128; w > 0; w >>= 1) {
        if (tid < w) red[tid] += red[tid + w];
        __syncthreads();
    }
    if (tid == 0) sum_prob[e] = red[0];
}

// =====================================================================
// 3. Offsets prefix-sum + aux loss
// =====================================================================
__global__ void moe_finalize(const int* __restrict__ counts, const float* __restrict__ sum_prob,
                             int* __restrict__ offsets, float* __restrict__ out_aux)
{
    if (threadIdx.x == 0 && blockIdx.x == 0) {
        int o = 0; float aux = 0.f;
        for (int e = 0; e < NEXP; ++e) {
            offsets[e] = o; o += counts[e];
            aux += ((float)counts[e] / (float)T_TOK) * (sum_prob[e] / (float)T_TOK);
        }
        offsets[NEXP] = o;
        out_aux[0] = (float)NEXP * aux;
    }
}

// =====================================================================
// 4. Scatter tokens into expert-sorted slots
// =====================================================================
__global__ __launch_bounds__(256)
void moe_scatter(const int* __restrict__ expert_idx, const int* __restrict__ rankb,
                 const float* __restrict__ coef2, const int* __restrict__ offsets,
                 int* __restrict__ slot_token, float* __restrict__ slot_coef,
                 int* __restrict__ token_slot)
{
    int t = blockIdx.x * 256 + threadIdx.x;
    if (t >= T_TOK) return;
#pragma unroll
    for (int j = 0; j < TOPKK; ++j) {
        int e = expert_idx[t * 2 + j];
        int s = offsets[e] + rankb[t * 2 + j];
        slot_token[s] = t;
        slot_coef[s]  = coef2[t * 2 + j];
        token_slot[t * 2 + j] = s;
    }
}

// =====================================================================
// 5. Grouped GEMM 1: h = silu(X @ w1[e]) * (X @ w3[e])   (bf16 WMMA)
// =====================================================================
__global__ __launch_bounds__(256)
void moe_gemm1(const float* __restrict__ x, const float* __restrict__ w1,
               const float* __restrict__ w3, const int* __restrict__ counts,
               const int* __restrict__ offsets, const int* __restrict__ slot_token,
               unsigned short* __restrict__ h)
{
    int e    = blockIdx.z;
    int rows = counts[e];
    int mt0  = blockIdx.y * BM;
    if (mt0 >= rows) return;
    int m_base = offsets[e] + mt0;
    int m_end  = offsets[e] + rows;
    int n0     = blockIdx.x * BN;

    __shared__ alignas(16) unsigned short la [BM * BKP];
    __shared__ alignas(16) unsigned short lb1[BN * BKP];
    __shared__ alignas(16) unsigned short lb3[BN * BKP];

    int tid  = threadIdx.x;
    int wave = tid >> 5;
    int wm   = (wave & 3) * 32;   // wave M offset in block tile
    int wn   = (wave >> 2) * 64;  // wave N offset

    // A staging assignment: 2 threads per row, 16 elems each
    int ar  = tid >> 1;
    int akc = (tid & 1) * 16;
    int am  = m_base + ar;
    int tok = slot_token[(am < m_end) ? am : m_base];
    const float* xrow = x + (size_t)tok * DDIM + akc;

    const float* w1e = w1 + (size_t)e * DDIM * HDIM;
    const float* w3e = w3 + (size_t)e * DDIM * HDIM;

    v8f acc1[2][4] = {};
    v8f acc3[2][4] = {};

    for (int kt = 0; kt < DDIM / BK; ++kt) {
        __syncthreads();
        { // A tile: fp32 gathered token rows -> bf16 LDS
            const float* p = xrow + kt * BK;
#pragma unroll
            for (int j = 0; j < 16; j += 4) {
                f32x4 q = *(const f32x4*)(p + j);
#pragma unroll
                for (int i = 0; i < 4; ++i)
                    la[ar * BKP + akc + j + i] = f2bf(q[i]);
            }
        }
        // B tiles: [BK][BN] slab of w1/w3, transposed into LDS [BN][BKP]
#pragma unroll
        for (int c = 0; c < 2; ++c) {
            int lin = tid * 2 + c;
            int kk  = lin >> 4;
            int ng  = (lin & 15) * 8;
            const float* p1 = w1e + (size_t)(kt * BK + kk) * HDIM + n0 + ng;
            const float* p3 = w3e + (size_t)(kt * BK + kk) * HDIM + n0 + ng;
            if (kt + 1 < DDIM / BK) {   // global_prefetch_b8 of next K-slab
                __builtin_prefetch(p1 + (size_t)BK * HDIM, 0, 1);
                __builtin_prefetch(p3 + (size_t)BK * HDIM, 0, 1);
            }
            f32x4 q1a = *(const f32x4*)p1, q1b = *(const f32x4*)(p1 + 4);
            f32x4 q3a = *(const f32x4*)p3, q3b = *(const f32x4*)(p3 + 4);
#pragma unroll
            for (int i = 0; i < 4; ++i) {
                lb1[(ng + i)     * BKP + kk] = f2bf(q1a[i]);
                lb1[(ng + 4 + i) * BKP + kk] = f2bf(q1b[i]);
                lb3[(ng + i)     * BKP + kk] = f2bf(q3a[i]);
                lb3[(ng + 4 + i) * BKP + kk] = f2bf(q3b[i]);
            }
        }
        __syncthreads();

        v16bf afrag[2];
#pragma unroll
        for (int mt = 0; mt < 2; ++mt)
            afrag[mt] = load_frag(frag_ptr(la, wm + mt * 16));
#pragma unroll
        for (int nt = 0; nt < 4; ++nt) {
            v16bf b1 = load_frag(frag_ptr(lb1, wn + nt * 16));
            v16bf b3 = load_frag(frag_ptr(lb3, wn + nt * 16));
#pragma unroll
            for (int mt = 0; mt < 2; ++mt) {
                acc1[mt][nt] = __builtin_amdgcn_wmma_f32_16x16x32_bf16(
                    false, afrag[mt], false, b1, (short)0, acc1[mt][nt], false, false);
                acc3[mt][nt] = __builtin_amdgcn_wmma_f32_16x16x32_bf16(
                    false, afrag[mt], false, b3, (short)0, acc3[mt][nt], false, false);
            }
        }
    }

    // epilogue: SwiGLU -> bf16 h[slot][H]
    int lane  = tid & 31;
    int col   = lane & 15;
    int rbase = (lane & 16) ? 8 : 0;
#pragma unroll
    for (int mt = 0; mt < 2; ++mt) {
#pragma unroll
        for (int nt = 0; nt < 4; ++nt) {
#pragma unroll
            for (int i = 0; i < 8; ++i) {
                int m = m_base + wm + mt * 16 + rbase + i;
                if (m < m_end) {
                    int n = n0 + wn + nt * 16 + col;
                    float a = acc1[mt][nt][i];
                    float g = a / (1.f + __expf(-a));       // silu
                    h[(size_t)m * HDIM + n] = f2bf(g * acc3[mt][nt][i]);
                }
            }
        }
    }
}

// =====================================================================
// 6. Grouped GEMM 2: partial = coef * (h @ w2[e])   (bf16 WMMA, f32 out)
//    A-tile staged by the Tensor Data Mover (tensor_load_to_lds):
//    2D tile 128 rows x 64B, tensor_dim0_stride = H, with hardware LDS
//    padding (16 DWORDs data + 4 DWORDs pad) reproducing the 80B rows
//    the WMMA fragment loader expects.
// =====================================================================
__global__ __launch_bounds__(256)
void moe_gemm2(const unsigned short* __restrict__ h, const float* __restrict__ w2,
               const int* __restrict__ counts, const int* __restrict__ offsets,
               const float* __restrict__ slot_coef, float* __restrict__ partial)
{
    int e    = blockIdx.z;
    int rows = counts[e];
    int mt0  = blockIdx.y * BM;
    if (mt0 >= rows) return;
    int m_base = offsets[e] + mt0;
    int m_end  = offsets[e] + rows;
    int n0     = blockIdx.x * BN;

    __shared__ alignas(16) unsigned short la[BM * BKP];
    __shared__ alignas(16) unsigned short lb[BN * BKP];

    int tid  = threadIdx.x;
    int wave = tid >> 5;
    int wm   = (wave & 3) * 32;
    int wn   = (wave >> 2) * 64;

#if HAVE_TDM
    // TDM descriptor group 1 (uniform across K-loop)
    i32x8 g1;
    g1[0] = (int)((1u << 16)       // data_size = 2 bytes
                | (1u << 20)       // pad_enable
                | (3u << 22)       // pad_interval: 16 DWORDs between pads
                | (3u << 25));     // pad_amount: 4 DWORDs (16B) -> 80B rows
    g1[1] = (int)((unsigned)(HDIM & 0xFFFF) << 16);   // tensor_dim0 lo16 @ bits 63:48
    g1[2] = (int)(((unsigned)(HDIM >> 16) & 0xFFFFu)
                | ((unsigned)(NSLOT & 0xFFFF) << 16)); // tensor_dim0 hi / tensor_dim1 lo
    g1[3] = (int)(((unsigned)(NSLOT >> 16) & 0xFFFFu)
                | ((unsigned)BK << 16));               // tensor_dim1 hi / tile_dim0
    g1[4] = (int)(unsigned)BM;                         // tile_dim1 (tile_dim2 = 0)
    g1[5] = (int)(unsigned)HDIM;                       // tensor_dim0_stride lo32
    g1[6] = 0;
    g1[7] = 0;
    unsigned int lds_off = (unsigned int)(uintptr_t)(void*)la;
    unsigned long long hbase = (unsigned long long)(uintptr_t)h
                             + 2ull * ((unsigned long long)(unsigned)m_base * HDIM);
#else
    int ar  = tid >> 1;
    int akc = (tid & 1) * 16;
    int am  = m_base + ar; if (am > NSLOT - 1) am = NSLOT - 1;
    const unsigned short* hrow = h + (size_t)am * HDIM + akc;
#endif

    const float* w2e = w2 + (size_t)e * HDIM * DDIM;

    v8f acc[2][4] = {};

    for (int kt = 0; kt < HDIM / BK; ++kt) {
        __syncthreads();
#if HAVE_TDM
        if (tid == 0) {
            unsigned long long ga = hbase + 2ull * (unsigned)(kt * BK);
            u32x4 g0;
            g0[0] = 1u;                                   // count=1, user descriptor
            g0[1] = lds_off;                              // LDS destination
            g0[2] = (unsigned int)(ga & 0xFFFFFFFFu);     // global_addr[31:0]
            g0[3] = (unsigned int)((ga >> 32) & 0x1FFFFFFu) | (2u << 30); // addr hi | type=2
            i32x4 z4 = {0, 0, 0, 0};
#if __clang_major__ >= 23
            i32x8 z8 = {0, 0, 0, 0, 0, 0, 0, 0};
            __builtin_amdgcn_tensor_load_to_lds(g0, g1, z4, z4, z8, 0);
#else
            __builtin_amdgcn_tensor_load_to_lds(g0, g1, z4, z4, 0);
#endif
        }
#else
        { // manual A tile: bf16 copy straight through
            const unsigned short* p = hrow + kt * BK;
            *(u32x4*)&la[ar * BKP + akc]     = *(const u32x4*)(p);
            *(u32x4*)&la[ar * BKP + akc + 8] = *(const u32x4*)(p + 8);
        }
#endif
#pragma unroll
        for (int c = 0; c < 2; ++c) {
            int lin = tid * 2 + c;
            int kk  = lin >> 4;
            int ng  = (lin & 15) * 8;
            const float* p2 = w2e + (size_t)(kt * BK + kk) * DDIM + n0 + ng;
            if (kt + 1 < HDIM / BK)
                __builtin_prefetch(p2 + (size_t)BK * DDIM, 0, 1);
            f32x4 qa = *(const f32x4*)p2, qb = *(const f32x4*)(p2 + 4);
#pragma unroll
            for (int i = 0; i < 4; ++i) {
                lb[(ng + i)     * BKP + kk] = f2bf(qa[i]);
                lb[(ng + 4 + i) * BKP + kk] = f2bf(qb[i]);
            }
        }
#if HAVE_TDM
        __builtin_amdgcn_s_wait_tensorcnt(0);   // issuing wave drains TDM; no-op elsewhere
#endif
        __syncthreads();

        v16bf afrag[2];
#pragma unroll
        for (int mt = 0; mt < 2; ++mt)
            afrag[mt] = load_frag(frag_ptr(la, wm + mt * 16));
#pragma unroll
        for (int nt = 0; nt < 4; ++nt) {
            v16bf b = load_frag(frag_ptr(lb, wn + nt * 16));
#pragma unroll
            for (int mt = 0; mt < 2; ++mt)
                acc[mt][nt] = __builtin_amdgcn_wmma_f32_16x16x32_bf16(
                    false, afrag[mt], false, b, (short)0, acc[mt][nt], false, false);
        }
    }

    int lane  = tid & 31;
    int col   = lane & 15;
    int rbase = (lane & 16) ? 8 : 0;
#pragma unroll
    for (int mt = 0; mt < 2; ++mt) {
#pragma unroll
        for (int nt = 0; nt < 4; ++nt) {
#pragma unroll
            for (int i = 0; i < 8; ++i) {
                int m = m_base + wm + mt * 16 + rbase + i;
                if (m < m_end) {
                    int n = n0 + wn + nt * 16 + col;
                    float cf = slot_coef[m];
                    partial[(size_t)m * DDIM + n] = cf * acc[mt][nt][i];
                }
            }
        }
    }
}

// =====================================================================
// 7. Deterministic combine: out[t] = partial[slot0] + partial[slot1]
// =====================================================================
__global__ __launch_bounds__(256)
void moe_combine(const float* __restrict__ partial, const int* __restrict__ token_slot,
                 float* __restrict__ out)
{
    int g    = blockIdx.x * 256 + threadIdx.x;
    int base = g * 4;
    int t = base / DDIM;
    int d = base - t * DDIM;
    int s0 = token_slot[t * 2 + 0];
    int s1 = token_slot[t * 2 + 1];
    f32x4 p0 = *(const f32x4*)(partial + (size_t)s0 * DDIM + d);
    f32x4 p1 = *(const f32x4*)(partial + (size_t)s1 * DDIM + d);
    f32x4 r  = p0 + p1;
    *(f32x4*)(out + base) = r;
}

// =====================================================================
extern "C" void kernel_launch(void* const* d_in, const int* in_sizes, int n_in,
                              void* d_out, int out_size, void* d_ws, size_t ws_size,
                              hipStream_t stream)
{
    (void)in_sizes; (void)n_in; (void)out_size; (void)ws_size;
    const float* x  = (const float*)d_in[0];
    const float* rw = (const float*)d_in[1];
    const float* w1 = (const float*)d_in[2];
    const float* w3 = (const float*)d_in[3];
    const float* w2 = (const float*)d_in[4];
    float* out = (float*)d_out;

    char* ws = (char*)d_ws;
    size_t off = 0;
    auto alloc = [&](size_t bytes) -> void* {
        void* p = ws + off;
        off = (off + bytes + 255) & ~(size_t)255;
        return p;
    };
    int*   counts     = (int*)  alloc(NEXP * 4);
    float* sum_prob   = (float*)alloc(NEXP * 4);
    int*   offsets    = (int*)  alloc((NEXP + 1) * 4);
    int*   expert_idx = (int*)  alloc((size_t)T_TOK * 2 * 4);
    int*   rankb      = (int*)  alloc((size_t)T_TOK * 2 * 4);
    float* coef2      = (float*)alloc((size_t)T_TOK * 2 * 4);
    int*   token_slot = (int*)  alloc((size_t)T_TOK * 2 * 4);
    float* prob       = (float*)alloc((size_t)T_TOK * NEXP * 4);
    int*   slot_token = (int*)  alloc((size_t)NSLOT * 4);
    float* slot_coef  = (float*)alloc((size_t)NSLOT * 4);
    unsigned short* hbuf = (unsigned short*)alloc((size_t)NSLOT * HDIM * 2);
    float* partial    = (float*)alloc((size_t)NSLOT * DDIM * 4);

    hipMemsetAsync(counts, 0, NEXP * sizeof(int), stream);

    moe_router     <<<T_TOK / 8, 256, 0, stream>>>(x, rw, counts, expert_idx, rankb, coef2, prob);
    moe_reduce_prob<<<NEXP, 256, 0, stream>>>(prob, sum_prob);
    moe_finalize   <<<1, 1, 0, stream>>>(counts, sum_prob, offsets, out + (size_t)T_TOK * DDIM);
    moe_scatter    <<<T_TOK / 256, 256, 0, stream>>>(expert_idx, rankb, coef2, offsets,
                                                     slot_token, slot_coef, token_slot);
    moe_gemm1<<<dim3(HDIM / BN, T_TOK / BM, NEXP), 256, 0, stream>>>(
        x, w1, w3, counts, offsets, slot_token, hbuf);
    moe_gemm2<<<dim3(DDIM / BN, T_TOK / BM, NEXP), 256, 0, stream>>>(
        hbuf, w2, counts, offsets, slot_coef, partial);
    moe_combine<<<(T_TOK * DDIM / 4) / 256, 256, 0, stream>>>(partial, token_slot, out);
}
